// NodeLayer_13580686589985
// MI455X (gfx1250) — matrix operand order repaired and checked by reading, not compiled, not run on hardware
//
#include <hip/hip_runtime.h>
#include <hip/hip_bf16.h>

typedef __attribute__((ext_vector_type(16))) _Float16 v16h;
typedef __attribute__((ext_vector_type(8)))  _Float16 v8h;
typedef __attribute__((ext_vector_type(8)))  float    v8f;

#define UNITS  128
#define HIDDEN 512
#define BATCH  16384
#define NTOT   (BATCH * UNITS)          // 2097152
#define T1C    1.0f
#define TOLC   0.01f
#define MINDT  0.1f
#define NSTEPS 24

// x5 coefficients
#define C51 ((float)(16.0/135.0))
#define C53 ((float)(6656.0/12825.0))
#define C54 ((float)(28561.0/56430.0))
#define C55 ((float)(-9.0/50.0))
#define C56 ((float)(2.0/55.0))
// x4 coefficients
#define C41 ((float)(25.0/216.0))
#define C43 ((float)(1408.0/2565.0))
#define C44 ((float)(2197.0/4104.0))
#define C45 (-0.2f)

__device__ __forceinline__ float4 ldf4(const float* __restrict__ p) {
    return *(const float4*)p;
}
__device__ __forceinline__ void ld8(const float* __restrict__ p, float f[8]) {
    const float4 u = *(const float4*)p;
    const float4 v = *(const float4*)(p + 4);
    f[0]=u.x; f[1]=u.y; f[2]=u.z; f[3]=u.w;
    f[4]=v.x; f[5]=v.y; f[6]=v.z; f[7]=v.w;
}
__device__ __forceinline__ void fma8(const float* __restrict__ p, float c, float f[8]) {
    const float4 u = *(const float4*)p;
    const float4 v = *(const float4*)(p + 4);
    f[0]+=c*u.x; f[1]+=c*u.y; f[2]+=c*u.z; f[3]+=c*u.w;
    f[4]+=c*v.x; f[5]+=c*v.y; f[6]+=c*v.z; f[7]+=c*v.w;
}

// ---------------------------------------------------------------------------
// One-time prep kernels
// ---------------------------------------------------------------------------

// Convert row-major f32 W (K x N) into WMMA B-fragment-swizzled f16:
// fragment (ktile,ntile) is 512 contiguous halfs: [lane][e], where
// element (lane,e) = W[ktile*32 + 8*(lane>=16) + (e&7) + 16*(e>=8)][ntile*16 + (lane&15)].
__global__ void node_swizzleW(const float* __restrict__ W,
                              _Float16* __restrict__ Wsw, int K, int N) {
    int tid = blockIdx.x * blockDim.x + threadIdx.x;
    if (tid >= K * N) return;
    int frag = tid >> 9;          // /512
    int r    = tid & 511;
    int lane = r >> 4;
    int e    = r & 15;
    int ntiles = N >> 4;
    int ktile  = frag / ntiles;
    int ntile  = frag % ntiles;
    int kk = ktile * 32 + ((lane >> 4) << 3) + (e & 7) + ((e >> 3) << 4);
    int nn = (ntile << 4) + (lane & 15);
    Wsw[tid] = (_Float16)W[kk * N + nn];
}

__global__ void node_init_scalars(float* __restrict__ scal) {
    scal[0] = 0.0f;   // t
    scal[1] = 0.1f;   // dt
    scal[2] = 0.0f;   // h
    ((unsigned*)scal)[3] = 0u; // err bits
}

__global__ __launch_bounds__(256) void node_copy4(const float* __restrict__ src,
                                                  float* __restrict__ dst) {
    int i = (blockIdx.x * 256 + threadIdx.x) * 4;
    *(float4*)(dst + i) = *(const float4*)(src + i);
}

// ---------------------------------------------------------------------------
// Per-step scalar kernels
// ---------------------------------------------------------------------------
__global__ void node_prestep(float* __restrict__ scal) {
    float t  = scal[0];
    float dt = scal[1];
    float h  = T1C - t;
    h = fminf(fmaxf(h, 0.0f), dt);
    scal[2] = h;
    ((unsigned*)scal)[3] = 0u;
}

__global__ void node_update_scalars(float* __restrict__ scal) {
    float t   = scal[0];
    float dt  = scal[1];
    float h   = scal[2];
    float err = __uint_as_float(((unsigned*)scal)[3]);
    bool done   = t >= (T1C - 1e-8f);
    bool accept = (err <= TOLC) || (h <= MINDT + 1e-8f);
    float t_n = accept ? (t + h) : t;
    float scale = 0.9f * powf(TOLC / (err + 1e-12f), 0.2f);
    scale = fminf(fmaxf(scale, 0.2f), 5.0f);
    float dt_new = fmaxf(dt * scale, MINDT);
    scal[0] = done ? t  : t_n;
    scal[1] = done ? dt : dt_new;
}

// ---------------------------------------------------------------------------
// GEMM1 (fused): hidden = relu( (x + h*sum(c_s * k_s)) @ W1 + b1 ) as f16
// M=16384 K=128 N=512. One wave -> 16 rows x 256 cols (16 accum fragments).
// B fragments preloaded in groups of 4 (own registers) so loads pipeline
// against WMMA issue instead of forcing s_wait_loadcnt 0 per WMMA.
// ---------------------------------------------------------------------------
template<int NS>
__global__ __launch_bounds__(256) void node_gemm1(
    const float* __restrict__ x,
    const float* __restrict__ ka, const float* __restrict__ kb,
    const float* __restrict__ kc, const float* __restrict__ kd,
    const float* __restrict__ ke,
    float c0, float c1, float c2, float c3, float c4,
    const float* __restrict__ scal,
    const _Float16* __restrict__ W1sw,
    const float* __restrict__ b1,
    _Float16* __restrict__ hidden)
{
    const int lane  = threadIdx.x & 31;
    const int wid   = threadIdx.x >> 5;
    const int gw    = blockIdx.x * 8 + wid;
    const int mtile = gw >> 1;
    const int nhalf = gw & 1;
    const int row   = mtile << 4;
    const int ncol0 = nhalf << 8;
    const int m  = lane & 15;
    const int kh = lane >> 4;

    const float h = scal[2];
    const float* ks[5] = { ka, kb, kc, kd, ke };
    const float  hc[5] = { h*c0, h*c1, h*c2, h*c3, h*c4 };

    v8f acc[16];
    #pragma unroll
    for (int i = 0; i < 16; ++i) {
        #pragma unroll
        for (int j = 0; j < 8; ++j) acc[i][j] = 0.0f;
    }

    const v16h* __restrict__ Wv = (const v16h*)W1sw;

    #pragma unroll
    for (int kt = 0; kt < 4; ++kt) {
        // A fragment: lane holds row m, K chunks [base..base+7] and [base+16..base+23]
        const int base = (row + m) * UNITS + kt * 32 + kh * 8;
        float f0[8], f1[8];
        ld8(x + base,      f0);
        ld8(x + base + 16, f1);
        #pragma unroll
        for (int s = 0; s < NS; ++s) {
            fma8(ks[s] + base,      hc[s], f0);
            fma8(ks[s] + base + 16, hc[s], f1);
        }
        v16h a;
        #pragma unroll
        for (int e = 0; e < 8; ++e) {
            a[e]     = (_Float16)f0[e];
            a[e + 8] = (_Float16)f1[e];
        }
        #pragma unroll
        for (int g = 0; g < 4; ++g) {
            v16h bf[4];
            #pragma unroll
            for (int j = 0; j < 4; ++j) {
                const int frag = kt * 32 + (nhalf << 4) + g * 4 + j;
                bf[j] = Wv[frag * 32 + lane];
            }
            #pragma unroll
            for (int j = 0; j < 4; ++j) {
                acc[g * 4 + j] = __builtin_amdgcn_wmma_f32_16x16x32_f16(
                    false, a, false, bf[j], (short)0, acc[g * 4 + j], false, false);
            }
        }
    }

    // epilogue: bias + relu + f16 store
    #pragma unroll
    for (int nt = 0; nt < 16; ++nt) {
        const int n = ncol0 + (nt << 4) + m;
        const float bias = b1[n];
        #pragma unroll
        for (int r = 0; r < 8; ++r) {
            float v = acc[nt][r] + bias;
            v = v > 0.0f ? v : 0.0f;
            hidden[(row + r + kh * 8) * HIDDEN + n] = (_Float16)v;
        }
    }
}

// ---------------------------------------------------------------------------
// GEMM2: kout = hidden(f16) @ W2 + b2   (M=16384 K=512 N=128)
// One wave -> 16 rows x 128 cols (8 accum fragments). All 8 B fragments of
// a K-tile are preloaded into distinct registers before the WMMA burst.
// ---------------------------------------------------------------------------
__global__ __launch_bounds__(256) void node_gemm2(
    const _Float16* __restrict__ hidden,
    const _Float16* __restrict__ W2sw,
    const float* __restrict__ b2,
    float* __restrict__ kout)
{
    const int lane = threadIdx.x & 31;
    const int wid  = threadIdx.x >> 5;
    const int gw   = blockIdx.x * 8 + wid;
    const int row  = gw << 4;
    const int m  = lane & 15;
    const int kh = lane >> 4;

    v8f acc[8];
    #pragma unroll
    for (int i = 0; i < 8; ++i) {
        #pragma unroll
        for (int j = 0; j < 8; ++j) acc[i][j] = 0.0f;
    }

    const v16h* __restrict__ Wv = (const v16h*)W2sw;

    for (int kt = 0; kt < 16; ++kt) {
        const _Float16* __restrict__ hp =
            hidden + (row + m) * HIDDEN + kt * 32 + kh * 8;
        v8h lo = *(const v8h*)hp;
        v8h hi = *(const v8h*)(hp + 16);
        v16h a = __builtin_shufflevector(lo, hi,
                                         0,1,2,3,4,5,6,7,8,9,10,11,12,13,14,15);
        v16h bf[8];
        #pragma unroll
        for (int nt = 0; nt < 8; ++nt) {
            bf[nt] = Wv[(kt * 8 + nt) * 32 + lane];   // ntiles = 128/16 = 8
        }
        #pragma unroll
        for (int nt = 0; nt < 8; ++nt) {
            acc[nt] = __builtin_amdgcn_wmma_f32_16x16x32_f16(
                false, a, false, bf[nt], (short)0, acc[nt], false, false);
        }
    }

    #pragma unroll
    for (int nt = 0; nt < 8; ++nt) {
        const int n = (nt << 4) + m;
        const float bias = b2[n];
        #pragma unroll
        for (int r = 0; r < 8; ++r) {
            kout[(row + r + kh * 8) * UNITS + n] = acc[nt][r] + bias;
        }
    }
}

// ---------------------------------------------------------------------------
// finalize: x5 = x + h*sum(b_i k_i); err = max|x5-x4| via wave-max + atomicMax
// ---------------------------------------------------------------------------
__global__ __launch_bounds__(256) void node_finalize(
    const float* __restrict__ x,
    const float* __restrict__ k1, const float* __restrict__ k3,
    const float* __restrict__ k4, const float* __restrict__ k5,
    const float* __restrict__ k6,
    float* __restrict__ scal,
    float* __restrict__ xnew)
{
    const int i = (blockIdx.x * 256 + threadIdx.x) * 4;
    const float h = scal[2];
    float4 xv = ldf4(x  + i);
    float4 a1 = ldf4(k1 + i);
    float4 a3 = ldf4(k3 + i);
    float4 a4 = ldf4(k4 + i);
    float4 a5 = ldf4(k5 + i);
    float4 a6 = ldf4(k6 + i);
    const float xs[4] = { xv.x, xv.y, xv.z, xv.w };
    const float v1[4] = { a1.x, a1.y, a1.z, a1.w };
    const float v3[4] = { a3.x, a3.y, a3.z, a3.w };
    const float v4[4] = { a4.x, a4.y, a4.z, a4.w };
    const float v5[4] = { a5.x, a5.y, a5.z, a5.w };
    const float v6[4] = { a6.x, a6.y, a6.z, a6.w };
    float out[4];
    float mx = 0.0f;
    #pragma unroll
    for (int j = 0; j < 4; ++j) {
        float d5 = C51*v1[j] + C53*v3[j] + C54*v4[j] + C55*v5[j] + C56*v6[j];
        float d4 = C41*v1[j] + C43*v3[j] + C44*v4[j] + C45*v5[j];
        float x5 = xs[j] + h * d5;
        float x4 = xs[j] + h * d4;
        mx = fmaxf(mx, fabsf(x5 - x4));
        out[j] = x5;
    }
    float4 o; o.x = out[0]; o.y = out[1]; o.z = out[2]; o.w = out[3];
    *(float4*)(xnew + i) = o;

    #pragma unroll
    for (int s = 16; s > 0; s >>= 1)
        mx = fmaxf(mx, __shfl_xor(mx, s, 32));
    if ((threadIdx.x & 31) == 0)
        atomicMax(((unsigned*)scal) + 3, __float_as_uint(mx));
}

// x = where(done, x, where(accept, xnew, x)) — condition is grid-uniform
__global__ __launch_bounds__(256) void node_update_x(
    float* __restrict__ x, const float* __restrict__ xnew,
    const float* __restrict__ scal)
{
    const float t   = scal[0];
    const float h   = scal[2];
    const float err = __uint_as_float(((const unsigned*)scal)[3]);
    const bool done   = t >= (T1C - 1e-8f);
    const bool accept = (err <= TOLC) || (h <= MINDT + 1e-8f);
    if (!done && accept) {
        const int i = (blockIdx.x * 256 + threadIdx.x) * 4;
        *(float4*)(x + i) = *(const float4*)(xnew + i);
    }
}

// ---------------------------------------------------------------------------
extern "C" void kernel_launch(void* const* d_in, const int* in_sizes, int n_in,
                              void* d_out, int out_size, void* d_ws, size_t ws_size,
                              hipStream_t stream) {
    (void)in_sizes; (void)n_in; (void)out_size; (void)ws_size;
    const float* x0 = (const float*)d_in[0];
    const float* W1 = (const float*)d_in[1];
    const float* b1 = (const float*)d_in[2];
    const float* W2 = (const float*)d_in[3];
    const float* b2 = (const float*)d_in[4];
    float* x = (float*)d_out;                    // live state lives in d_out

    char* ws = (char*)d_ws;
    size_t off = 0;
    auto alloc = [&](size_t bytes) -> void* {
        void* p = ws + off;
        off = (off + bytes + 255) & ~(size_t)255;
        return p;
    };
    float* scal = (float*)alloc(256);            // [t, dt, h, err_bits]
    float* kbuf[6];
    for (int i = 0; i < 6; ++i) kbuf[i] = (float*)alloc((size_t)NTOT * 4);
    _Float16* hidden = (_Float16*)alloc((size_t)BATCH * HIDDEN * 2);
    _Float16* W1sw   = (_Float16*)alloc((size_t)UNITS * HIDDEN * 2);
    _Float16* W2sw   = (_Float16*)alloc((size_t)HIDDEN * UNITS * 2);
    float* xnew = (float*)hidden;                // reuse: hidden dead by finalize

    const int BLK4 = NTOT / 4 / 256;             // 2048 blocks for elementwise
    const int G1B  = (BATCH / 16) * 2 / 8;       // 256 blocks for gemm1
    const int G2B  = (BATCH / 16) / 8;           // 128 blocks for gemm2

    // one-time prep (runs every call: deterministic)
    node_swizzleW<<<(UNITS * HIDDEN + 255) / 256, 256, 0, stream>>>(W1, W1sw, UNITS, HIDDEN);
    node_swizzleW<<<(HIDDEN * UNITS + 255) / 256, 256, 0, stream>>>(W2, W2sw, HIDDEN, UNITS);
    node_init_scalars<<<1, 1, 0, stream>>>(scal);
    node_copy4<<<BLK4, 256, 0, stream>>>(x0, x);

    for (int step = 0; step < NSTEPS; ++step) {
        node_prestep<<<1, 1, 0, stream>>>(scal);

        // k1 = f(x)
        node_gemm1<0><<<G1B, 256, 0, stream>>>(x, kbuf[0], kbuf[0], kbuf[0], kbuf[0], kbuf[0],
            0.f, 0.f, 0.f, 0.f, 0.f, scal, W1sw, b1, hidden);
        node_gemm2<<<G2B, 256, 0, stream>>>(hidden, W2sw, b2, kbuf[0]);
        // k2 = f(x + h*(1/4 k1))
        node_gemm1<1><<<G1B, 256, 0, stream>>>(x, kbuf[0], kbuf[0], kbuf[0], kbuf[0], kbuf[0],
            0.25f, 0.f, 0.f, 0.f, 0.f, scal, W1sw, b1, hidden);
        node_gemm2<<<G2B, 256, 0, stream>>>(hidden, W2sw, b2, kbuf[1]);
        // k3 = f(x + h*(3/32 k1 + 9/32 k2))
        node_gemm1<2><<<G1B, 256, 0, stream>>>(x, kbuf[0], kbuf[1], kbuf[0], kbuf[0], kbuf[0],
            (float)(3.0/32.0), (float)(9.0/32.0), 0.f, 0.f, 0.f, scal, W1sw, b1, hidden);
        node_gemm2<<<G2B, 256, 0, stream>>>(hidden, W2sw, b2, kbuf[2]);
        // k4
        node_gemm1<3><<<G1B, 256, 0, stream>>>(x, kbuf[0], kbuf[1], kbuf[2], kbuf[0], kbuf[0],
            (float)(1932.0/2197.0), (float)(-7200.0/2197.0), (float)(7296.0/2197.0),
            0.f, 0.f, scal, W1sw, b1, hidden);
        node_gemm2<<<G2B, 256, 0, stream>>>(hidden, W2sw, b2, kbuf[3]);
        // k5
        node_gemm1<4><<<G1B, 256, 0, stream>>>(x, kbuf[0], kbuf[1], kbuf[2], kbuf[3], kbuf[0],
            (float)(439.0/216.0), -8.0f, (float)(3680.0/513.0), (float)(-845.0/4104.0),
            0.f, scal, W1sw, b1, hidden);
        node_gemm2<<<G2B, 256, 0, stream>>>(hidden, W2sw, b2, kbuf[4]);
        // k6
        node_gemm1<5><<<G1B, 256, 0, stream>>>(x, kbuf[0], kbuf[1], kbuf[2], kbuf[3], kbuf[4],
            (float)(-8.0/27.0), 2.0f, (float)(-3544.0/2565.0), (float)(1859.0/4104.0),
            (float)(-11.0/40.0), scal, W1sw, b1, hidden);
        node_gemm2<<<G2B, 256, 0, stream>>>(hidden, W2sw, b2, kbuf[5]);

        node_finalize<<<BLK4, 256, 0, stream>>>(x, kbuf[0], kbuf[2], kbuf[3], kbuf[4], kbuf[5],
                                                scal, xnew);
        node_update_x<<<BLK4, 256, 0, stream>>>(x, xnew, scal);
        node_update_scalars<<<1, 1, 0, stream>>>(scal);
    }
}
